// HyperSpectralClassifier_51445118272196
// MI455X (gfx1250) — matrix operand order repaired and checked by reading, compiled, not run on hardware
//
#include <hip/hip_runtime.h>
#include <hip/hip_bf16.h>

typedef __attribute__((ext_vector_type(16))) _Float16 v16h;
typedef __attribute__((ext_vector_type(8)))  float    v8f;

__device__ __forceinline__ float gelu_exact(float x) {
  return 0.5f * x * (1.0f + erff(x * 0.7071067811865475f));
}

// ---------------------------------------------------------------------------
// Generic strided WMMA GEMM:  Out[m][n] = alpha * sum_k A[m,k] * B[n,k] (+bias[n])
// A element:  A + z*aB + m*sAm + k*sAk
// B element:  B + z*bB + n*sBn + k*sBk
// Out:        O + z*oB + m*sOm + n*sOn     (accumulate==1 -> +=)
// One wave (32 lanes) per 16x16 tile, K stepped by 32, f16 in / f32 acc.
// Compile-time specialization FA/FB: K-stride-1 + 16B-aligned operands use
// float4 (global_load_b128) fetch; otherwise strided scalar fetch. Main loop
// is bound-check free; tail uses clamped addresses + post-load select.
// ---------------------------------------------------------------------------
template <bool FA, bool FB>
__global__ void k_gemm_wmma(const float* __restrict__ A, const float* __restrict__ B,
                            const float* __restrict__ bias, float* __restrict__ O,
                            int M, int N, int K,
                            long long sAm, long long sAk, long long aB,
                            long long sBn, long long sBk, long long bB,
                            long long sOm, long long sOn, long long oB,
                            float alpha, int accumulate) {
  const int lane = threadIdx.x & 31;     // wave32
  const int l16  = lane & 15;
  const int hi   = (lane >> 4) & 1;
  const int tm   = blockIdx.x * 16;
  const int tn   = blockIdx.y * 16;
  const float* Ab = A + (long long)blockIdx.z * aB;
  const float* Bb = B + (long long)blockIdx.z * bB;
  float*       Ob = O + (long long)blockIdx.z * oB;

  int am = tm + l16; if (am >= M) am = M - 1;   // clamp loads, guard stores
  int bn = tn + l16; if (bn >= N) bn = N - 1;
  const float* Arow = Ab + (long long)am * sAm;
  const float* Brow = Bb + (long long)bn * sBn;

  v8f acc = {0.f,0.f,0.f,0.f,0.f,0.f,0.f,0.f};
  const int kfull = K & ~31;

#pragma unroll 2
  for (int k0 = 0; k0 < kfull; k0 += 32) {
    v16h av, bv;
    // ---- A operand: elements 0..7 -> K = k0+hi*8+e ; 8..15 -> K = k0+16+hi*8+(e-8)
    if (FA) {
      const float4* pa0 = (const float4*)(Arow + k0 + hi * 8);
      const float4* pa1 = (const float4*)(Arow + k0 + 16 + hi * 8);
      float4 a0 = pa0[0], a1 = pa0[1], a2 = pa1[0], a3 = pa1[1];
      av[0]=(_Float16)a0.x;  av[1]=(_Float16)a0.y;  av[2]=(_Float16)a0.z;  av[3]=(_Float16)a0.w;
      av[4]=(_Float16)a1.x;  av[5]=(_Float16)a1.y;  av[6]=(_Float16)a1.z;  av[7]=(_Float16)a1.w;
      av[8]=(_Float16)a2.x;  av[9]=(_Float16)a2.y;  av[10]=(_Float16)a2.z; av[11]=(_Float16)a2.w;
      av[12]=(_Float16)a3.x; av[13]=(_Float16)a3.y; av[14]=(_Float16)a3.z; av[15]=(_Float16)a3.w;
    } else {
#pragma unroll
      for (int e = 0; e < 16; ++e) {
        const int koff = (e < 8) ? (hi * 8 + e) : (16 + hi * 8 + (e - 8));
        av[e] = (_Float16)Arow[(long long)(k0 + koff) * sAk];
      }
    }
    // ---- B operand: elements 0..7 -> K = k0+hi*8+e ; 8..15 -> K = k0+16+hi*8+(e-8)
    if (FB) {
      const float4* pb = (const float4*)(Brow + k0 + hi * 8);
      const float4* pb1 = (const float4*)(Brow + k0 + 16 + hi * 8);
      float4 b0 = pb[0], b1 = pb[1], b2 = pb1[0], b3 = pb1[1];
      bv[0]=(_Float16)b0.x;  bv[1]=(_Float16)b0.y;  bv[2]=(_Float16)b0.z;  bv[3]=(_Float16)b0.w;
      bv[4]=(_Float16)b1.x;  bv[5]=(_Float16)b1.y;  bv[6]=(_Float16)b1.z;  bv[7]=(_Float16)b1.w;
      bv[8]=(_Float16)b2.x;  bv[9]=(_Float16)b2.y;  bv[10]=(_Float16)b2.z; bv[11]=(_Float16)b2.w;
      bv[12]=(_Float16)b3.x; bv[13]=(_Float16)b3.y; bv[14]=(_Float16)b3.z; bv[15]=(_Float16)b3.w;
    } else {
#pragma unroll
      for (int e = 0; e < 16; ++e) {
        bv[e] = (_Float16)Brow[(long long)(k0 + ((e < 8) ? (hi * 8 + e) : (16 + hi * 8 + (e - 8)))) * sBk];
      }
    }
    acc = __builtin_amdgcn_wmma_f32_16x16x32_f16(false, av, false, bv,
                                                 (short)0, acc, false, false);
  }

  // ---- K tail (< 32): clamped addresses + post-load zero select, no branches
  if (kfull < K) {
    const int Km1 = K - 1;
    v16h av, bv;
#pragma unroll
    for (int e = 0; e < 16; ++e) {
      const int koff = (e < 8) ? (hi * 8 + e) : (16 + hi * 8 + (e - 8));
      const int ka = kfull + koff;
      const int kac = (ka < K) ? ka : Km1;
      float fa = Arow[(long long)kac * sAk];
      av[e] = (_Float16)((ka < K) ? fa : 0.0f);
      const int kb = kfull + koff;
      const int kbc = (kb < K) ? kb : Km1;
      float fb = Brow[(long long)kbc * sBk];
      bv[e] = (_Float16)((kb < K) ? fb : 0.0f);
    }
    acc = __builtin_amdgcn_wmma_f32_16x16x32_f16(false, av, false, bv,
                                                 (short)0, acc, false, false);
  }

  // C/D layout: VGPR r -> row m = tm + r + 8*hi, col n = tn + l16
#pragma unroll
  for (int r = 0; r < 8; ++r) {
    const int m = tm + r + hi * 8;
    const int n = tn + l16;
    if (m < M && n < N) {
      float val = acc[r] * alpha;
      if (bias) val += bias[n];
      const long long off = (long long)m * sOm + (long long)n * sOn;
      if (accumulate) Ob[off] += val; else Ob[off] = val;
    }
  }
}

// ---------------------------------------------------------------------------
// Direct 3x3 conv, zero padding=1, arbitrary stride. NCHW.
// ---------------------------------------------------------------------------
__global__ void k_conv3x3(const float* __restrict__ in, const float* __restrict__ w,
                          const float* __restrict__ bias, float* __restrict__ out,
                          int Bz, int Cin, int Cout, int Hin, int Win,
                          int Hout, int Wout, int stride) {
  int idx = blockIdx.x * blockDim.x + threadIdx.x;
  int total = Bz * Cout * Hout * Wout;
  if (idx >= total) return;
  int wo = idx % Wout; int t = idx / Wout;
  int ho = t % Hout; t /= Hout;
  int co = t % Cout; int b = t / Cout;
  float acc = bias[co];
  for (int ci = 0; ci < Cin; ++ci) {
    const float* ip = in + ((long long)(b * Cin + ci)) * Hin * Win;
    const float* wp = w  + ((long long)(co * Cin + ci)) * 9;
    for (int kh = 0; kh < 3; ++kh) {
      int hh = ho * stride - 1 + kh;
      if (hh < 0 || hh >= Hin) continue;
      for (int kw = 0; kw < 3; ++kw) {
        int ww = wo * stride - 1 + kw;
        if (ww < 0 || ww >= Win) continue;
        acc += ip[hh * Win + ww] * wp[kh * 3 + kw];
      }
    }
  }
  out[idx] = acc;
}

// Depthwise 3x3 conv (zero pad=1, stride 1) fused with exact GELU.
__global__ void k_dwconv3x3_gelu(const float* __restrict__ in, const float* __restrict__ w,
                                 const float* __restrict__ bias, float* __restrict__ out,
                                 int Bz, int C, int H, int W) {
  int idx = blockIdx.x * blockDim.x + threadIdx.x;
  int total = Bz * C * H * W;
  if (idx >= total) return;
  int x = idx % W; int t = idx / W;
  int y = t % H; t /= H;
  int c = t % C; int b = t / C;
  const float* ip = in + ((long long)(b * C + c)) * H * W;
  const float* wp = w + c * 9;
  float acc = bias[c];
  for (int kh = 0; kh < 3; ++kh) {
    int hh = y - 1 + kh; if (hh < 0 || hh >= H) continue;
    for (int kw = 0; kw < 3; ++kw) {
      int ww = x - 1 + kw; if (ww < 0 || ww >= W) continue;
      acc += ip[hh * W + ww] * wp[kh * 3 + kw];
    }
  }
  out[idx] = gelu_exact(acc);
}

// Eval BatchNorm (per channel) + exact GELU. Layout [Z, C, HW].
__global__ void k_bn_gelu(const float* __restrict__ in, float* __restrict__ out,
                          const float* __restrict__ g, const float* __restrict__ b,
                          const float* __restrict__ m, const float* __restrict__ v,
                          int C, int HW, int total) {
  int idx = blockIdx.x * blockDim.x + threadIdx.x;
  if (idx >= total) return;
  int c = (idx / HW) % C;
  float x = (in[idx] - m[c]) * rsqrtf(v[c] + 1e-5f) * g[c] + b[c];
  out[idx] = gelu_exact(x);
}

// Global average pool over HW; one thread per (b,c).
__global__ void k_gap(const float* __restrict__ in, float* __restrict__ out,
                      int HW, int total_bc) {
  int idx = blockIdx.x * blockDim.x + threadIdx.x;
  if (idx >= total_bc) return;
  const float* p = in + (long long)idx * HW;
  float s = 0.f;
  for (int i = 0; i < HW; ++i) s += p[i];
  out[idx] = s / (float)HW;
}

// Small fully-connected: out[b][o] = act(W[o,:] . in[b,:] + bias[o])
__global__ void k_fc(const float* __restrict__ in, const float* __restrict__ w,
                     const float* __restrict__ bias, float* __restrict__ out,
                     int Bz, int I, int O, int do_gelu) {
  int idx = blockIdx.x * blockDim.x + threadIdx.x;
  if (idx >= Bz * O) return;
  int o = idx % O, b = idx / O;
  const float* ip = in + (long long)b * I;
  const float* wp = w + (long long)o * I;
  float acc = bias[o];
  for (int i = 0; i < I; ++i) acc += wp[i] * ip[i];
  out[idx] = do_gelu ? gelu_exact(acc) : acc;
}

// Dynamic per-channel 3x3 conv with reflect padding (pad=1), weights w9[b,c,9].
__global__ void k_local_dyn(const float* __restrict__ x, const float* __restrict__ w9,
                            float* __restrict__ out, int Bz, int C, int H, int W) {
  int idx = blockIdx.x * blockDim.x + threadIdx.x;
  int total = Bz * C * H * W;
  if (idx >= total) return;
  int xx = idx % W; int t = idx / W;
  int yy = t % H; t /= H;
  int c = t % C; int b = t / C;
  const float* ip = x + ((long long)(b * C + c)) * H * W;
  const float* wp = w9 + ((long long)(b * C + c)) * 9;
  float acc = 0.f;
  for (int kh = 0; kh < 3; ++kh) {
    int h = yy - 1 + kh;
    if (h < 0) h = -h; if (h >= H) h = 2 * H - 2 - h;   // reflect
    for (int kw = 0; kw < 3; ++kw) {
      int w_ = xx - 1 + kw;
      if (w_ < 0) w_ = -w_; if (w_ >= W) w_ = 2 * W - 2 - w_;
      acc += ip[h * W + w_] * wp[kh * 3 + kw];
    }
  }
  out[idx] = acc;
}

// LayerNorm over last dim C, in place on [T, C] buffer. blockDim must be 256.
__global__ void k_layernorm(float* __restrict__ t, const float* __restrict__ g,
                            const float* __restrict__ b, int C) {
  float* p = t + (long long)blockIdx.x * C;
  __shared__ float red[256];
  float s = 0.f;
  for (int c = threadIdx.x; c < C; c += 256) s += p[c];
  red[threadIdx.x] = s; __syncthreads();
  for (int st = 128; st > 0; st >>= 1) { if (threadIdx.x < st) red[threadIdx.x] += red[threadIdx.x + st]; __syncthreads(); }
  float mu = red[0] / (float)C; __syncthreads();
  float v = 0.f;
  for (int c = threadIdx.x; c < C; c += 256) { float d = p[c] - mu; v += d * d; }
  red[threadIdx.x] = v; __syncthreads();
  for (int st = 128; st > 0; st >>= 1) { if (threadIdx.x < st) red[threadIdx.x] += red[threadIdx.x + st]; __syncthreads(); }
  float inv = rsqrtf(red[0] / (float)C + 1e-5f);
  __syncthreads();
  for (int c = threadIdx.x; c < C; c += 256) p[c] = (p[c] - mu) * inv * g[c] + b[c];
}

// Row softmax in place. blockDim must be 256. One block per row.
__global__ void k_softmax_rows(float* __restrict__ d, int cols) {
  float* p = d + (long long)blockIdx.x * cols;
  __shared__ float red[256];
  float m = -3.0e38f;
  for (int c = threadIdx.x; c < cols; c += 256) m = fmaxf(m, p[c]);
  red[threadIdx.x] = m; __syncthreads();
  for (int st = 128; st > 0; st >>= 1) { if (threadIdx.x < st) red[threadIdx.x] = fmaxf(red[threadIdx.x], red[threadIdx.x + st]); __syncthreads(); }
  m = red[0]; __syncthreads();
  float s = 0.f;
  for (int c = threadIdx.x; c < cols; c += 256) { float e = expf(p[c] - m); p[c] = e; s += e; }
  red[threadIdx.x] = s; __syncthreads();
  for (int st = 128; st > 0; st >>= 1) { if (threadIdx.x < st) red[threadIdx.x] += red[threadIdx.x + st]; __syncthreads(); }
  float inv = 1.0f / red[0];
  __syncthreads();
  for (int c = threadIdx.x; c < cols; c += 256) p[c] *= inv;
}

// Top-1 MoE gate: softmax over E logits, gate[argmax] = max/(max+1e-8).
__global__ void k_moe_gate(const float* __restrict__ logits, float* __restrict__ gates,
                           int Bz, int E) {
  int b = blockIdx.x * blockDim.x + threadIdx.x;
  if (b >= Bz) return;
  const float* l = logits + b * E;
  float* g = gates + b * E;
  float m = -3.0e38f;
  for (int e = 0; e < E; ++e) m = fmaxf(m, l[e]);
  float s = 0.f;
  for (int e = 0; e < E; ++e) s += expf(l[e] - m);
  int arg = 0; float best = -1.f;
  for (int e = 0; e < E; ++e) {
    float r = expf(l[e] - m) / s;
    g[e] = 0.f;
    if (r > best) { best = r; arg = e; }
  }
  g[arg] = best / (best + 1e-8f);
}

__global__ void k_copy(const float* __restrict__ src, float* __restrict__ dst, int total) {
  int idx = blockIdx.x * blockDim.x + threadIdx.x;
  if (idx < total) dst[idx] = src[idx];
}

// dst += gates[b*E+e] * src   over layout [B, CHW]
__global__ void k_axpy_gate(const float* __restrict__ src, float* __restrict__ dst,
                            const float* __restrict__ gates, int e, int E,
                            int CHW, int total) {
  int idx = blockIdx.x * blockDim.x + threadIdx.x;
  if (idx >= total) return;
  int b = idx / CHW;
  dst[idx] += gates[b * E + e] * src[idx];
}

// ---------------------------------------------------------------------------
struct BlockP {
  const float *dbn_b,*dbn_g,*dbn_m,*dbn_v,*down_b,*down_w;
  const float *fc_b,*fc_w,*in_b,*in_w,*ln_b,*ln_g,*out_b,*out_w;
  const float *dn1_b,*dn1_w,*dn2_b,*dn2_w;
  const float *dw_b,*dw_w,*pw_b,*pw_w,*r_b,*r_w;
  const float *pbn_b,*pbn_g,*pbn_m,*pbn_v,*proj_b,*proj_w;
};

extern "C" void kernel_launch(void* const* d_in, const int* in_sizes, int n_in,
                              void* d_out, int out_size, void* d_ws, size_t ws_size,
                              hipStream_t stream) {
  (void)in_sizes; (void)out_size;
  if (n_in < 131) return;  // pytree-flattened param count + x

  // --- Input pointer walk. Assumes JAX pytree (alphabetical-key) flatten:
  // 'params' before 'x'; within params: blocks, bn1_*, head_*, pre1_*, pre2_*.
  int ii = 0;
  auto nxt = [&]() { return (const float*)d_in[ii++]; };
  BlockP bp[4];
  for (int i = 0; i < 4; ++i) {
    BlockP& p = bp[i];
    p.dbn_b = nxt(); p.dbn_g = nxt(); p.dbn_m = nxt(); p.dbn_v = nxt();
    p.down_b = nxt(); p.down_w = nxt();
    p.fc_b = nxt(); p.fc_w = nxt(); p.in_b = nxt(); p.in_w = nxt();
    p.ln_b = nxt(); p.ln_g = nxt(); p.out_b = nxt(); p.out_w = nxt();
    p.dn1_b = nxt(); p.dn1_w = nxt(); p.dn2_b = nxt(); p.dn2_w = nxt();
    p.dw_b = nxt(); p.dw_w = nxt(); p.pw_b = nxt(); p.pw_w = nxt();
    p.r_b = nxt(); p.r_w = nxt();
    p.pbn_b = nxt(); p.pbn_g = nxt(); p.pbn_m = nxt(); p.pbn_v = nxt();
    p.proj_b = nxt(); p.proj_w = nxt();
  }
  const float* bn1_b = nxt(); const float* bn1_g = nxt();
  const float* bn1_m = nxt(); const float* bn1_v = nxt();
  const float* head_b = nxt(); const float* head_w = nxt();
  const float* pre1_b = nxt(); const float* pre1_w = nxt();
  const float* pre2_b = nxt(); const float* pre2_w = nxt();
  const float* x_in = nxt();

  // --- Workspace carve (all float32) ---
  char* wsb = (char*)d_ws;
  size_t off = 0;
  auto alloc = [&](size_t nfloat) -> float* {
    float* p = (float*)(wsb + off);
    off += ((nfloat * 4 + 255) / 256) * 256;
    return p;
  };
  float* xbuf   = alloc(262144);   // current activation, NCHW
  float* fused  = alloc(262144);
  float* tbuf   = alloc(262144);   // tokens [T, C]
  float* qkvbuf = alloc(786432);   // [T, 3C]
  float* sc     = alloc(1048576);  // one (b,h) score tile, S x S
  float* obuf   = alloc(262144);   // attention out tokens [T, C]
  float* tmp1   = alloc(262144);
  float* tmp2   = alloc(131072);
  float* fbuf   = alloc(262144);
  float* dwbuf  = alloc(262144);
  float* pwbuf  = alloc(262144);
  float* xnew   = alloc(262144);
  float* gbuf   = alloc(2048);
  float* hbuf   = alloc(2048);
  float* w9buf  = alloc(18432);
  float* logits = alloc(16);
  float* gates  = alloc(16);
  if (off > ws_size) return;  // insufficient scratch; nothing safe to do

  // Dispatch one of four compile-time GEMM specializations based on K-stride
  // and 16B alignment (checked for every z-slice via the batch stride).
  auto gemm = [&](const float* A, const float* B, const float* bias, float* O,
                  int M, int N, int K,
                  long long sAm, long long sAk, long long aB,
                  long long sBn, long long sBk, long long bB,
                  long long sOm, long long sOn, long long oB,
                  int Z, float alpha, int acc) {
    dim3 grid((M + 15) / 16, (N + 15) / 16, Z);
    const bool fA = (sAk == 1) && ((sAm & 3LL) == 0) && ((aB & 3LL) == 0) &&
                    ((((size_t)A) & 15) == 0);
    const bool fB = (sBk == 1) && ((sBn & 3LL) == 0) && ((bB & 3LL) == 0) &&
                    ((((size_t)B) & 15) == 0);
    if (fA && fB)
      k_gemm_wmma<true, true><<<grid, dim3(32), 0, stream>>>(A, B, bias, O, M, N, K,
          sAm, sAk, aB, sBn, sBk, bB, sOm, sOn, oB, alpha, acc);
    else if (fA)
      k_gemm_wmma<true, false><<<grid, dim3(32), 0, stream>>>(A, B, bias, O, M, N, K,
          sAm, sAk, aB, sBn, sBk, bB, sOm, sOn, oB, alpha, acc);
    else if (fB)
      k_gemm_wmma<false, true><<<grid, dim3(32), 0, stream>>>(A, B, bias, O, M, N, K,
          sAm, sAk, aB, sBn, sBk, bB, sOm, sOn, oB, alpha, acc);
    else
      k_gemm_wmma<false, false><<<grid, dim3(32), 0, stream>>>(A, B, bias, O, M, N, K,
          sAm, sAk, aB, sBn, sBk, bB, sOm, sOn, oB, alpha, acc);
  };
  auto blk = [](int n) { return (n + 255) / 256; };

  // ================= Stage 0 =================
  // pre1: 1x1 conv 200->30 as GEMM  (M=HW, N=Cout, K=Cin, per-batch z)
  gemm(x_in, pre1_w, pre1_b, tmp1,
       1024, 30, 200,
       1LL, 1024LL, 200LL * 1024,
       200LL, 1LL, 0LL,
       1LL, 1024LL, 30LL * 1024,
       4, 1.0f, 0);
  int tot = 4 * 30 * 1024;
  k_bn_gelu<<<blk(tot), 256, 0, stream>>>(tmp1, tmp2, bn1_g, bn1_b, bn1_m, bn1_v, 30, 1024, tot);
  tot = 4 * 64 * 1024;
  k_conv3x3<<<blk(tot), 256, 0, stream>>>(tmp2, pre2_w, pre2_b, xbuf,
                                          4, 30, 64, 32, 32, 32, 32, 1);

  // ================= Blocks =================
  for (int i = 0; i < 4; ++i) {
    const BlockP& p = bp[i];
    const int d = 64 << i, H = 32 >> i, W = H, HW = H * W, S = HW, T = 4 * S;
    const int hd = d / 8, d2 = 2 * d, Ho = H / 2;
    tot = 4 * d * HW;

    // ---- local unit: GAP -> dn1(gelu) -> dn2 -> dynamic 3x3 (reflect) ----
    k_gap<<<blk(4 * d), 256, 0, stream>>>(xbuf, gbuf, HW, 4 * d);
    k_fc<<<blk(4 * d), 256, 0, stream>>>(gbuf, p.dn1_w, p.dn1_b, hbuf, 4, d, d, 1);
    k_fc<<<blk(4 * 9 * d), 256, 0, stream>>>(hbuf, p.dn2_w, p.dn2_b, w9buf, 4, d, 9 * d, 0);
    k_local_dyn<<<blk(tot), 256, 0, stream>>>(xbuf, w9buf, fused, 4, d, H, W);

    // ---- global unit ----
    // token fc: tok[s][c] = sum_k x[b,k,s] * fc_w[c,k] + fc_b   (rfft/irfft = identity)
    gemm(xbuf, p.fc_w, p.fc_b, tbuf,
         S, d, d,
         1LL, (long long)HW, (long long)d * HW,
         (long long)d, 1LL, 0LL,
         (long long)d, 1LL, (long long)S * d,
         4, 1.0f, 0);
    k_layernorm<<<T, 256, 0, stream>>>(tbuf, p.ln_g, p.ln_b, d);
    // qkv: [T, 3d]
    gemm(tbuf, p.in_w, p.in_b, qkvbuf,
         T, 3 * d, d,
         (long long)d, 1LL, 0LL,
         (long long)d, 1LL, 0LL,
         (long long)(3 * d), 1LL, 0LL,
         1, 1.0f, 0);
    const float scale = rsqrtf((float)hd);
    for (int b = 0; b < 4; ++b) {
      for (int h = 0; h < 8; ++h) {
        const float* qb = qkvbuf + (long long)b * S * 3 * d + h * hd;
        const float* kb = qb + d;
        const float* vb = qb + 2 * d;
        // scores[s][t] = (q_s . k_t) * scale
        gemm(qb, kb, nullptr, sc,
             S, S, hd,
             (long long)(3 * d), 1LL, 0LL,
             (long long)(3 * d), 1LL, 0LL,
             (long long)S, 1LL, 0LL,
             1, scale, 0);
        k_softmax_rows<<<S, 256, 0, stream>>>(sc, S);
        // o[s][e] = sum_t attn[s][t] * v[t][e]
        gemm(sc, vb, nullptr, obuf + (long long)b * S * d + h * hd,
             S, hd, S,
             (long long)S, 1LL, 0LL,
             1LL, (long long)(3 * d), 0LL,
             (long long)d, 1LL, 0LL,
             1, 1.0f, 0);
      }
    }
    // out proj, accumulate into fused (fused = local + global), NCHW output
    gemm(obuf, p.out_w, p.out_b, fused,
         S, d, d,
         (long long)d, 1LL, (long long)S * d,
         (long long)d, 1LL, 0LL,
         1LL, (long long)HW, (long long)d * HW,
         4, 1.0f, 1);

    // ---- proj 1x1 conv + BN + GELU ----
    gemm(fused, p.proj_w, p.proj_b, tmp1,
         S, d, d,
         1LL, (long long)HW, (long long)d * HW,
         (long long)d, 1LL, 0LL,
         1LL, (long long)HW, (long long)d * HW,
         4, 1.0f, 0);
    k_bn_gelu<<<blk(tot), 256, 0, stream>>>(tmp1, fbuf, p.pbn_g, p.pbn_b, p.pbn_m, p.pbn_v, d, HW, tot);

    // ---- MoE (top-1 of 4 experts) + residual ----
    k_gap<<<blk(4 * d), 256, 0, stream>>>(fbuf, gbuf, HW, 4 * d);
    k_fc<<<1, 256, 0, stream>>>(gbuf, p.r_w, p.r_b, logits, 4, d, 4, 0);
    k_moe_gate<<<1, 32, 0, stream>>>(logits, gates, 4, 4);
    k_copy<<<blk(tot), 256, 0, stream>>>(xbuf, xnew, tot);   // residual init
    for (int e = 0; e < 4; ++e) {
      k_dwconv3x3_gelu<<<blk(tot), 256, 0, stream>>>(fbuf, p.dw_w + (long long)e * d * 9,
                                                     p.dw_b + e * d, dwbuf, 4, d, H, W);
      gemm(dwbuf, p.pw_w + (long long)e * d * d, p.pw_b + e * d, pwbuf,
           S, d, d,
           1LL, (long long)HW, (long long)d * HW,
           (long long)d, 1LL, 0LL,
           1LL, (long long)HW, (long long)d * HW,
           4, 1.0f, 0);
      k_axpy_gate<<<blk(tot), 256, 0, stream>>>(pwbuf, xnew, gates, e, 4, d * HW, tot);
    }

    // ---- downsample conv 3x3 stride2 + BN + GELU ----
    const int tot2 = 4 * d2 * Ho * Ho;
    k_conv3x3<<<blk(tot2), 256, 0, stream>>>(xnew, p.down_w, p.down_b, tmp1,
                                             4, d, d2, H, W, Ho, Ho, 2);
    k_bn_gelu<<<blk(tot2), 256, 0, stream>>>(tmp1, xbuf, p.dbn_g, p.dbn_b, p.dbn_m, p.dbn_v,
                                             d2, Ho * Ho, tot2);
  }

  // ================= Head =================
  k_gap<<<blk(4 * 512), 256, 0, stream>>>(xbuf, gbuf, 16, 4 * 512);
  gemm(gbuf, head_w, head_b, (float*)d_out,
       4, 16, 512,
       512LL, 1LL, 0LL,
       512LL, 1LL, 0LL,
       16LL, 1LL, 0LL,
       1, 1.0f, 0);
}
